// GatedSparseAttention_25640954757719
// MI455X (gfx1250) — matrix operand order repaired and checked by reading, not compiled, not run on hardware
//
#include <hip/hip_runtime.h>
#include <hip/hip_bf16.h>
#include <math.h>

typedef __bf16 bf16_t;
typedef __attribute__((ext_vector_type(16))) __bf16 v16bf;
typedef __attribute__((ext_vector_type(8)))  float  v8f;
typedef unsigned int u32x4 __attribute__((ext_vector_type(4)));
typedef int i32x4 __attribute__((ext_vector_type(4)));
typedef int i32x8 __attribute__((ext_vector_type(8)));

__device__ __forceinline__ bf16_t f2bf(float f) {
  unsigned u = __float_as_uint(f);
  unsigned r = (u + 0x7FFFu + ((u >> 16) & 1u)) >> 16;
  unsigned short s = (unsigned short)r;
  return __builtin_bit_cast(bf16_t, s);
}
__device__ __forceinline__ float sigmoidf_(float x) { return 1.0f / (1.0f + __expf(-x)); }

// ---------------------------------------------------------------------------
// C[M,N] = A[M,K] * W[N,K]^T   (both row-major, K contiguous) via bf16 WMMA
// Block: 256 threads = 8 waves; tile 64(M) x 64(N); K step 32.
// ---------------------------------------------------------------------------
#define TK 32
__global__ void __launch_bounds__(256)
gemm_bf16_nt(const float* __restrict__ A, const float* __restrict__ W,
             float* __restrict__ C, int M, int N, int K) {
  __shared__ bf16_t As[64][TK];
  __shared__ bf16_t Bs[64][TK];
  const int tid  = threadIdx.x;
  const int lane = tid & 31;
  const int wv   = tid >> 5;
  const int m0 = blockIdx.y * 64, n0 = blockIdx.x * 64;
  const int mtile = wv & 3;   // 0..3
  const int npair = wv >> 2;  // 0..1 -> n-tiles {2*npair, 2*npair+1}
  const int hh = lane >> 4, nl = lane & 15;
  v8f c0 = {}, c1 = {};
  for (int k0 = 0; k0 < K; k0 += TK) {
    if (k0 + TK < K) { // prefetch next K-tile (global_prefetch_b8)
      __builtin_prefetch(&A[(size_t)(m0 + (tid & 63)) * K + k0 + TK], 0, 3);
      __builtin_prefetch(&W[(size_t)(n0 + (tid & 63)) * K + k0 + TK], 0, 3);
    }
    for (int i = tid; i < 64 * TK; i += 256) {
      int r = i >> 5, kk = i & 31;
      As[r][kk] = f2bf(A[(size_t)(m0 + r) * K + k0 + kk]);
      Bs[r][kk] = f2bf(W[(size_t)(n0 + r) * K + k0 + kk]);
    }
    __syncthreads();
    v16bf a, b0, b1;
#pragma unroll
    for (int e = 0; e < 16; ++e) {
      int ka = 8 * hh + (e & 7) + 16 * (e >> 3);  // A-fragment K index
      int kb = 8 * hh + (e & 7) + 16 * (e >> 3);  // B-fragment K index
      a[e]  = As[mtile * 16 + nl][ka];
      b0[e] = Bs[(npair * 2 + 0) * 16 + nl][kb];
      b1[e] = Bs[(npair * 2 + 1) * 16 + nl][kb];
    }
    c0 = __builtin_amdgcn_wmma_f32_16x16x32_bf16(false, a, false, b0, (short)0, c0, false, false);
    c1 = __builtin_amdgcn_wmma_f32_16x16x32_bf16(false, a, false, b1, (short)0, c1, false, false);
    __syncthreads();
  }
#pragma unroll
  for (int r = 0; r < 8; ++r) {
    int m = m0 + mtile * 16 + r + 8 * hh;
    if (m < M) {
      C[(size_t)m * N + n0 + (npair * 2 + 0) * 16 + nl] = c0[r];
      C[(size_t)m * N + n0 + (npair * 2 + 1) * 16 + nl] = c1[r];
    }
  }
}

// ---------------------------------------------------------------------------
// wv[t,h] = sigmoid(x[t,:] . W_Iw[h,:])   (T x 4 tiny GEMV)
// ---------------------------------------------------------------------------
__global__ void indexer_w_kernel(const float* __restrict__ x,
                                 const float* __restrict__ W_Iw,
                                 float* __restrict__ wvout, int T, int C) {
  int idx = blockIdx.x * blockDim.x + threadIdx.x;  // t*4 + h
  if (idx >= T * 4) return;
  int t = idx >> 2, h = idx & 3;
  const float* xr = x + (size_t)t * C;
  const float* wr = W_Iw + (size_t)h * C;
  float acc = 0.f;
  for (int i = 0; i < C; ++i) acc += xr[i] * wr[i];
  wvout[idx] = sigmoidf_(acc);
}

// ---------------------------------------------------------------------------
// RoPE in place, exact reference index permutation (hd = 64).
// ---------------------------------------------------------------------------
__global__ void rope_kernel(float* __restrict__ qk, int T) {
  int idx = blockIdx.x * blockDim.x + threadIdx.x;  // t*16 + head
  if (idx >= T * 16) return;
  int t = idx >> 4, h = idx & 15;
  float* p = qk + (size_t)t * 1024 + h * 64;
  float xv[64];
#pragma unroll
  for (int i = 0; i < 64; ++i) xv[i] = p[i];
#pragma unroll
  for (int j = 0; j < 32; ++j) {
    int fi = (2 * j) & 31;
    float ang = (float)t * powf(10000.0f, -((float)(2 * fi)) / 64.0f);
    float ce = cosf(ang), se = sinf(ang);
    float x1 = xv[2 * j], x2 = xv[2 * j + 1];
    p[j]      = x1 * ce - x2 * se;
    p[32 + j] = x1 * se + x2 * ce;
  }
}

__global__ void gate_mul_kernel(float* __restrict__ a, const float* __restrict__ g, int n) {
  int i = blockIdx.x * blockDim.x + threadIdx.x;
  if (i < n) a[i] = a[i] * sigmoidf_(g[i]);
}

// ---------------------------------------------------------------------------
// Per (head, query): importance row -> variance -> k_t -> exact top-k_t
// selection bitmask. Radix MSB-first k-th largest on float bit patterns.
// ---------------------------------------------------------------------------
__global__ void __launch_bounds__(256)
select_kernel(const float* __restrict__ qI, const float* __restrict__ kI,
              const float* __restrict__ wv, const float* __restrict__ gate_bias,
              const float* __restrict__ head_bias, unsigned* __restrict__ mask, int T) {
  const int qpos = blockIdx.x;
  const int h    = blockIdx.y;
  const int hi   = h >> 2;  // jnp.repeat(., 4, axis=1)
  const int tid  = threadIdx.x;
  const int lane = tid & 31, wvid = tid >> 5;
  __shared__ unsigned uimp[2048];
  __shared__ float qrow[32];
  __shared__ float redf[8], redf2[8];
  __shared__ unsigned redu[8];
  __shared__ int scanbuf[256];
  __shared__ int selflag[256];
  __shared__ int sh_misc[2];

  if (tid < 32) qrow[tid] = qI[(size_t)qpos * 128 + hi * 32 + tid];
  __syncthreads();
  const float gb  = gate_bias[hi];
  const float sib = sigmoidf_(head_bias[h]);
  const float wq  = wv[qpos * 4 + hi];
  float sum = 0.f, sumsq = 0.f;
  for (int k = tid; k < T; k += 256) {
    if (k <= qpos) {
      const float* kr = kI + (size_t)k * 128 + hi * 32;
      float d = 0.f;
#pragma unroll
      for (int i = 0; i < 32; ++i) d += qrow[i] * kr[i];
      float val = sib * wq * sigmoidf_(d + gb);
      sum += val; sumsq += val * val;                        // variance: no inf
      float vs = (k < 4) ? __int_as_float(0x7f800000) : val; // sinks = +inf
      uimp[k] = __float_as_uint(vs);
    } else {
      uimp[k] = 0u;
    }
  }
  for (int o = 16; o; o >>= 1) { sum += __shfl_xor(sum, o, 32); sumsq += __shfl_xor(sumsq, o, 32); }
  if (lane == 0) { redf[wvid] = sum; redf2[wvid] = sumsq; }
  __syncthreads();
  if (tid == 0) {
    float s = 0.f, s2 = 0.f;
    for (int i = 0; i < 8; ++i) { s += redf[i]; s2 += redf2[i]; }
    float mean = s / (float)T;
    float var  = s2 / (float)T - mean * mean;
    int kt = (int)floorf(512.0f * var);
    if (kt < 32) kt = 32;
    if (kt > 1024) kt = 1024;
    sh_misc[0] = kt;
  }
  __syncthreads();
  const int kt = sh_misc[0];
  const int ncausal = qpos + 1;
  unsigned* mrow = mask + ((size_t)h * T + qpos) * (T / 32);

  if (ncausal <= kt) {
    for (int wj = tid; wj < T / 32; wj += 256) {
      unsigned bits = 0u;
      int kb = wj * 32;
      for (int b = 0; b < 32; ++b)
        if (kb + b <= qpos) bits |= (1u << b);
      mrow[wj] = bits;
    }
    return;
  }
  unsigned thr = 0u;
  for (int b = 30; b >= 0; --b) {
    unsigned cand = thr | (1u << b);
    int cnt = 0;
    for (int k = tid; k < T; k += 256) cnt += (uimp[k] >= cand) ? 1 : 0;
    for (int o = 16; o; o >>= 1) cnt += __shfl_xor(cnt, o, 32);
    if (lane == 0) redu[wvid] = (unsigned)cnt;
    __syncthreads();
    if (tid == 0) { int c = 0; for (int i = 0; i < 8; ++i) c += (int)redu[i]; sh_misc[1] = c; }
    __syncthreads();
    if (sh_misc[1] >= kt) thr = cand;
    __syncthreads();
  }
  {
    int cnt = 0;
    for (int k = tid; k < T; k += 256) cnt += (uimp[k] > thr) ? 1 : 0;
    for (int o = 16; o; o >>= 1) cnt += __shfl_xor(cnt, o, 32);
    if (lane == 0) redu[wvid] = (unsigned)cnt;
    __syncthreads();
    if (tid == 0) { int c = 0; for (int i = 0; i < 8; ++i) c += (int)redu[i]; sh_misc[1] = c; }
    __syncthreads();
  }
  const int cgt = sh_misc[1];
  int rties = kt - cgt; if (rties < 0) rties = 0;
  int running = 0;
  for (int c = 0; c < T / 256; ++c) {
    int k = c * 256 + tid;
    unsigned u = uimp[k];
    int tie = (u == thr) ? 1 : 0;
    int gt  = (u > thr) ? 1 : 0;
    scanbuf[tid] = tie;
    __syncthreads();
    for (int off = 1; off < 256; off <<= 1) {
      int tmp = (tid >= off) ? scanbuf[tid - off] : 0;
      __syncthreads();
      scanbuf[tid] += tmp;
      __syncthreads();
    }
    int excl = scanbuf[tid] - tie;
    int sel  = gt | (tie && (running + excl) < rties);
    selflag[tid] = sel;
    __syncthreads();
    if (tid < 8) {
      unsigned bits = 0u;
      for (int b = 0; b < 32; ++b) bits |= (selflag[tid * 32 + b] ? 1u : 0u) << b;
      mrow[c * 8 + tid] = bits;
    }
    running += scanbuf[255];
    __syncthreads();
  }
}

// ---------------------------------------------------------------------------
// Flash-style masked attention: one wave per (head, 16-query tile).
// K chunk staged by the Tensor Data Mover; V chunk staged by async loads.
// ---------------------------------------------------------------------------
__global__ void __launch_bounds__(32)
attn_kernel(const float* __restrict__ q, const float* __restrict__ k,
            const float* __restrict__ v, const unsigned* __restrict__ mask,
            float* __restrict__ o, int T) {
  const int qt = blockIdx.x;
  const int h  = blockIdx.y;
  const int lane = threadIdx.x;
  const int hh = lane >> 4, nl = lane & 15;
  __shared__ __align__(16) float vbuf[32][64];
  __shared__ __align__(16) float kbuf[32][64];
  __shared__ bf16_t pbuf[16][32];
  __shared__ unsigned mrow[16];

  v16bf aq0, aq1;  // Q tile A-fragments (K = dims 0..31 / 32..63)
  {
    const float* qr = q + (size_t)(qt * 16 + nl) * 1024 + h * 64;
#pragma unroll
    for (int e = 0; e < 16; ++e) {
      int d = 8 * hh + (e & 7) + 16 * (e >> 3);
      aq0[e] = f2bf(qr[d]);
      aq1[e] = f2bf(qr[d + 32]);
    }
  }
  float m[8], s[8];
#pragma unroll
  for (int r = 0; r < 8; ++r) { m[r] = -3.0e38f; s[r] = 0.0f; }
  v8f o0 = {}, o1 = {}, o2 = {}, o3 = {};
  const int qmax = qt * 16 + 15;

  for (int kb = 0; kb <= qmax; kb += 32) {
    // ---- TDM: K chunk (32 keys x 64 dims, fp32, row stride 1024) -> kbuf ----
    {
      unsigned ldsoff = (unsigned)(size_t)&kbuf[0][0];  // low 32 bits of flat = LDS offset
      unsigned long long ga = (unsigned long long)(size_t)(k + (size_t)kb * 1024 + h * 64);
      u32x4 g0 = { 1u,                                   // count=1 (user D#)
                   ldsoff,                               // lds_addr
                   (unsigned)(ga & 0xFFFFFFFFull),       // global_addr lo
                   (unsigned)((ga >> 32) & 0x1FFFFFFull) | (2u << 30) };  // hi + type=2
      i32x8 g1 = { (int)(2u << 16),      // data_size = 4B
                   (int)(1024u << 16),   // tensor_dim0 = 1024 (lo16)
                   (int)(2048u << 16),   // tensor_dim0 hi = 0 | tensor_dim1 lo16 = 2048
                   (int)(64u << 16),     // tensor_dim1 hi = 0 | tile_dim0 = 64
                   32,                   // tile_dim1 = 32 | tile_dim2 = 0
                   1024,                 // tensor_dim0_stride lo32 = 1024
                   0, 0 };
      i32x4 gz = { 0, 0, 0, 0 };
#if __has_builtin(__builtin_amdgcn_tensor_load_to_lds)
# if defined(__clang_major__) && (__clang_major__ >= 23)
      i32x8 gx = { 0, 0, 0, 0, 0, 0, 0, 0 };
      __builtin_amdgcn_tensor_load_to_lds(g0, g1, gz, gz, gx, 0);
# else
      __builtin_amdgcn_tensor_load_to_lds(g0, g1, gz, gz, 0);
# endif
#else
      for (int i = lane; i < 32 * 64; i += 32) {
        int rr = i >> 6, dd = i & 63;
        kbuf[rr][dd] = k[(size_t)(kb + rr) * 1024 + h * 64 + dd];
      }
#endif
    }
    // ---- async loads: V chunk (32 keys x 64 dims) -> vbuf, 16B granules ----
    {
      unsigned vl = (unsigned)(size_t)&vbuf[0][0];
      const float* vg = v + (size_t)kb * 1024 + h * 64;
      for (int i = lane; i < 512; i += 32) {  // 512 float4 groups
        int rr = i >> 4, dd = (i & 15) * 4;
        unsigned long long ga = (unsigned long long)(size_t)(vg + (size_t)rr * 1024 + dd);
        unsigned ldsa = vl + (unsigned)((rr * 64 + dd) * 4);
        asm volatile("global_load_async_to_lds_b128 %0, %1, off"
                     :: "v"(ldsa), "v"(ga) : "memory");
      }
    }
    if (lane < 16)
      mrow[lane] = mask[((size_t)h * T + qt * 16 + lane) * (T / 32) + (kb >> 5)];
    asm volatile("s_wait_asynccnt 0x0" ::: "memory");
#if __has_builtin(__builtin_amdgcn_s_wait_tensorcnt)
    __builtin_amdgcn_s_wait_tensorcnt(0);
#else
    asm volatile("s_wait_tensorcnt 0x0" ::: "memory");
#endif
    __syncthreads();

    v8f sct[2];
#pragma unroll
    for (int st = 0; st < 2; ++st) {  // two 16-key score tiles from LDS K
      v16bf b0, b1;
#pragma unroll
      for (int e = 0; e < 16; ++e) {
        int d = 8 * hh + (e & 7) + 16 * (e >> 3);
        b0[e] = f2bf(kbuf[st * 16 + nl][d]);
        b1[e] = f2bf(kbuf[st * 16 + nl][d + 32]);
      }
      v8f sc = {};
      sc = __builtin_amdgcn_wmma_f32_16x16x32_bf16(false, aq0, false, b0, (short)0, sc, false, false);
      sc = __builtin_amdgcn_wmma_f32_16x16x32_bf16(false, aq1, false, b1, (short)0, sc, false, false);
#pragma unroll
      for (int r = 0; r < 8; ++r) {
        float val = sc[r] * 0.125f;  // 1/sqrt(64)
        unsigned bits = mrow[r + 8 * hh];
        if (!((bits >> (nl + st * 16)) & 1u)) val = -3.0e38f;
        sc[r] = val;
      }
      sct[st] = sc;
    }
    // online softmax over the full 32-key chunk
#pragma unroll
    for (int r = 0; r < 8; ++r) {
      float rm = fmaxf(sct[0][r], sct[1][r]);
#pragma unroll
      for (int off = 1; off < 16; off <<= 1) rm = fmaxf(rm, __shfl_xor(rm, off, 32));
      float mn = fmaxf(m[r], rm);
      float cf = __expf(m[r] - mn);
      float p0 = __expf(sct[0][r] - mn);
      float p1 = __expf(sct[1][r] - mn);
      float ps = p0 + p1;
#pragma unroll
      for (int off = 1; off < 16; off <<= 1) ps += __shfl_xor(ps, off, 32);
      s[r] = s[r] * cf + ps;
      m[r] = mn;
      o0[r] *= cf; o1[r] *= cf; o2[r] *= cf; o3[r] *= cf;
      pbuf[r + 8 * hh][nl]      = f2bf(p0);
      pbuf[r + 8 * hh][16 + nl] = f2bf(p1);
    }
    __syncthreads();
    // O += P(16x32) * V(32x64)
    v16bf ap, bv0, bv1, bv2, bv3;
#pragma unroll
    for (int e = 0; e < 16; ++e) {
      int kc = 8 * hh + (e & 7) + 16 * (e >> 3);
      ap[e] = pbuf[nl][kc];
      int ki = 8 * hh + (e & 7) + 16 * (e >> 3);
      bv0[e] = f2bf(vbuf[ki][0 * 16 + nl]);
      bv1[e] = f2bf(vbuf[ki][1 * 16 + nl]);
      bv2[e] = f2bf(vbuf[ki][2 * 16 + nl]);
      bv3[e] = f2bf(vbuf[ki][3 * 16 + nl]);
    }
    o0 = __builtin_amdgcn_wmma_f32_16x16x32_bf16(false, ap, false, bv0, (short)0, o0, false, false);
    o1 = __builtin_amdgcn_wmma_f32_16x16x32_bf16(false, ap, false, bv1, (short)0, o1, false, false);
    o2 = __builtin_amdgcn_wmma_f32_16x16x32_bf16(false, ap, false, bv2, (short)0, o2, false, false);
    o3 = __builtin_amdgcn_wmma_f32_16x16x32_bf16(false, ap, false, bv3, (short)0, o3, false, false);
    __syncthreads();
  }
#pragma unroll
  for (int r = 0; r < 8; ++r) {
    int t = qt * 16 + r + 8 * hh;
    float inv = 1.0f / s[r];
    float* orow = o + (size_t)t * 1024 + h * 64;
    orow[0 * 16 + nl] = o0[r] * inv;
    orow[1 * 16 + nl] = o1[r] * inv;
    orow[2 * 16 + nl] = o2[r] * inv;
    orow[3 * 16 + nl] = o3[r] * inv;
  }
}

// ---------------------------------------------------------------------------
extern "C" void kernel_launch(void* const* d_in, const int* in_sizes, int n_in,
                              void* d_out, int out_size, void* d_ws, size_t ws_size,
                              hipStream_t stream) {
  (void)in_sizes; (void)n_in; (void)out_size; (void)ws_size;
  const float* x         = (const float*)d_in[0];
  const float* W_Iq      = (const float*)d_in[1];
  const float* W_Ik      = (const float*)d_in[2];
  const float* W_Iw      = (const float*)d_in[3];
  const float* gate_bias = (const float*)d_in[4];
  const float* head_bias = (const float*)d_in[5];
  const float* W_q       = (const float*)d_in[6];
  const float* W_k       = (const float*)d_in[7];
  const float* W_v       = (const float*)d_in[8];
  const float* W_gv      = (const float*)d_in[9];
  const float* W_go      = (const float*)d_in[10];
  const float* W_o       = (const float*)d_in[11];
  float* out = (float*)d_out;

  const int T = 2048, C = 1024;
  float* ws = (float*)d_ws;
  size_t off = 0;
  float* qI = ws + off; off += (size_t)T * 128;
  float* kI = ws + off; off += (size_t)T * 128;
  float* wv = ws + off; off += (size_t)T * 4;
  float* qb = ws + off; off += (size_t)T * C;
  float* kb = ws + off; off += (size_t)T * C;
  float* vb = ws + off; off += (size_t)T * C;
  float* gv = ws + off; off += (size_t)T * C;   // reused as o after attention
  float* go = ws + off; off += (size_t)T * C;
  unsigned* selmask = (unsigned*)(ws + off);    // 16*T*(T/32) words = 8 MB

  dim3 blk(256);
  gemm_bf16_nt<<<dim3(128 / 64, T / 64), blk, 0, stream>>>(x, W_Iq, qI, T, 128, C);
  gemm_bf16_nt<<<dim3(128 / 64, T / 64), blk, 0, stream>>>(x, W_Ik, kI, T, 128, C);
  indexer_w_kernel<<<(T * 4 + 255) / 256, blk, 0, stream>>>(x, W_Iw, wv, T, C);
  gemm_bf16_nt<<<dim3(C / 64, T / 64), blk, 0, stream>>>(x, W_q, qb, T, C, C);
  gemm_bf16_nt<<<dim3(C / 64, T / 64), blk, 0, stream>>>(x, W_k, kb, T, C, C);
  gemm_bf16_nt<<<dim3(C / 64, T / 64), blk, 0, stream>>>(x, W_v, vb, T, C, C);
  gemm_bf16_nt<<<dim3(C / 64, T / 64), blk, 0, stream>>>(x, W_gv, gv, T, C, C);
  gemm_bf16_nt<<<dim3(C / 64, T / 64), blk, 0, stream>>>(x, W_go, go, T, C, C);
  gate_mul_kernel<<<(T * C + 255) / 256, blk, 0, stream>>>(vb, gv, T * C);
  rope_kernel<<<(T * 16 + 255) / 256, blk, 0, stream>>>(qb, T);
  rope_kernel<<<(T * 16 + 255) / 256, blk, 0, stream>>>(kb, T);
  select_kernel<<<dim3(T, 16), blk, 0, stream>>>(qI, kI, wv, gate_bias, head_bias, selmask, T);
  float* ob = gv;
  attn_kernel<<<dim3(T / 16, 16), dim3(32), 0, stream>>>(qb, kb, vb, selmask, ob, T);
  gate_mul_kernel<<<(T * C + 255) / 256, blk, 0, stream>>>(ob, go, T * C);
  gemm_bf16_nt<<<dim3(C / 64, T / 64), blk, 0, stream>>>(ob, W_o, out, T, C, C);
}
